// HardTripletLoss_29446295781455
// MI455X (gfx1250) — compile-verified, matrix operand order checked by reading, not run on hardware
//
#include <hip/hip_runtime.h>
#include <hip/hip_bf16.h>

// Problem constants (from reference)
constexpr int GS  = 8;      // GRID_SIZE
constexpr int NB  = 4;      // B
constexpr int NN  = 1024;   // N
constexpr int NC  = 256;    // C
constexpr int NH  = 60;     // H
constexpr int NW  = 80;     // W
constexpr int NHW = NH * NW; // 4800

typedef __attribute__((ext_vector_type(16))) __bf16          v16bf;
typedef __attribute__((ext_vector_type(8)))  float           v8f;
typedef __attribute__((ext_vector_type(8)))  unsigned short  u16x8;

static __device__ __forceinline__ unsigned short f2bf(float f) {
  unsigned int u = __float_as_uint(f);
  u += 0x7FFFu + ((u >> 16) & 1u);           // round-to-nearest-even
  return (unsigned short)(u >> 16);
}

static __device__ __forceinline__ float wred32(float v) {
  #pragma unroll
  for (int o = 16; o > 0; o >>= 1) v += __shfl_xor(v, o, 32);  // wave32
  return v;
}

// Branchless sorted-insert of v into ascending (q0<=q1<=q2<=q3): keeps 4 smallest.
static __device__ __forceinline__ void min4_insert(float v, float& q0, float& q1,
                                                   float& q2, float& q3) {
  q3 = fminf(q3, fmaxf(q2, v));
  q2 = fminf(q2, fmaxf(q1, v));
  q1 = fminf(q1, fmaxf(q0, v));
  q0 = fminf(q0, v);
}

__global__ void zero_out_kernel(float* out) {
  if (blockIdx.x == 0 && threadIdx.x == 0) out[0] = 0.0f;
}

// One wave (32 lanes) per keypoint row: bilinear sample + L2-normalize + pos
__global__ void prep_kd_pos_kernel(const float* __restrict__ kp1_desc,
                                   const float* __restrict__ w_kp1,
                                   const float* __restrict__ desc2,
                                   unsigned short* __restrict__ kdbf,
                                   float* __restrict__ posb) {
  const int wave = threadIdx.x >> 5;
  const int lane = threadIdx.x & 31;
  const int row  = blockIdx.x * 8 + wave;      // b*NN + n
  if (row >= NB * NN) return;
  const int b = row / NN;

  const float px = w_kp1[row * 2 + 0] * (1.0f / GS) - 0.5f;
  const float py = w_kp1[row * 2 + 1] * (1.0f / GS) - 0.5f;
  const float x0 = floorf(px), y0 = floorf(py);
  const float wx = px - x0,   wy = py - y0;
  const int x0i = min(max((int)x0, 0), NW - 1);
  const int x1i = min(x0i + 1, NW - 1);
  const int y0i = min(max((int)y0, 0), NH - 1);
  const int y1i = min(y0i + 1, NH - 1);

  float kk = 0.0f, ww = 0.0f, kw = 0.0f;
  float kvals[8];
  #pragma unroll
  for (int i = 0; i < 8; ++i) {
    const int c = lane * 8 + i;
    const float kv = kp1_desc[(size_t)row * NC + c];
    const float* base = desc2 + (size_t)(b * NC + c) * NHW;
    const float d00 = base[y0i * NW + x0i];
    const float d01 = base[y0i * NW + x1i];
    const float d10 = base[y1i * NW + x0i];
    const float d11 = base[y1i * NW + x1i];
    const float top = d00 * (1.0f - wx) + d01 * wx;
    const float bot = d10 * (1.0f - wx) + d11 * wx;
    const float wv  = top * (1.0f - wy) + bot * wy;
    kvals[i] = kv;
    kk += kv * kv; ww += wv * wv; kw += kv * wv;
  }
  kk = wred32(kk); ww = wred32(ww); kw = wred32(kw);
  const float invk = 1.0f / sqrtf(kk + 1e-8f);
  const float invw = 1.0f / sqrtf(ww + 1e-8f);
  #pragma unroll
  for (int i = 0; i < 8; ++i)
    kdbf[(size_t)row * NC + lane * 8 + i] = f2bf(kvals[i] * invk);
  if (lane == 0) posb[row] = 2.0f - 2.0f * kw * invk * invw;
}

// One wave per (b, hw) pixel: transpose + L2-normalize desc2 -> bf16 [b][hw][c]
__global__ void prep_d2f_kernel(const float* __restrict__ desc2,
                                unsigned short* __restrict__ d2fbf) {
  const int wave = threadIdx.x >> 5;
  const int lane = threadIdx.x & 31;
  const int row  = blockIdx.x * 8 + wave;      // b*NHW + m
  if (row >= NB * NHW) return;
  const int b = row / NHW;
  const int m = row - b * NHW;

  float ss = 0.0f;
  float vals[8];
  #pragma unroll
  for (int i = 0; i < 8; ++i) {
    const int c = lane * 8 + i;
    const float v = desc2[(size_t)(b * NC + c) * NHW + m];
    vals[i] = v; ss += v * v;
  }
  ss = wred32(ss);
  const float inv = 1.0f / sqrtf(ss + 1e-8f);
  #pragma unroll
  for (int i = 0; i < 8; ++i)
    d2fbf[(size_t)row * NC + lane * 8 + i] = f2bf(vals[i] * inv);
}

// Top-4 nearest grid-cell centers over a clamped 5x5 window (contains truth).
static __device__ __forceinline__ void top4cells(float px, float py, int* out) {
  const float fx = px * (1.0f / GS) - 0.5f;
  const float fy = py * (1.0f / GS) - 0.5f;
  int ix0 = min(max((int)floorf(fx) - 2, 0), NW - 5);
  int iy0 = min(max((int)floorf(fy) - 2, 0), NH - 5);
  float d0 = 3.0e38f, d1 = 3.0e38f, d2 = 3.0e38f, d3 = 3.0e38f;
  int   i0 = 0, i1 = 0, i2 = 0, i3 = 0;
  for (int yy = 0; yy < 5; ++yy) {
    for (int xx = 0; xx < 5; ++xx) {
      const int ix = ix0 + xx, iy = iy0 + yy;
      const float dx = (ix + 0.5f) * GS - px;
      const float dy = (iy + 0.5f) * GS - py;
      const float d  = dx * dx + dy * dy;
      const int   m  = iy * NW + ix;
      if (d < d3) {
        d3 = d; i3 = m;
        if (d3 < d2) { float t = d2; d2 = d3; d3 = t; int u = i2; i2 = i3; i3 = u; }
        if (d2 < d1) { float t = d1; d1 = d2; d2 = t; int u = i1; i1 = i2; i2 = u; }
        if (d1 < d0) { float t = d0; d0 = d1; d1 = t; int u = i0; i0 = i1; i1 = u; }
      }
    }
  }
  out[0] = i0; out[1] = i1; out[2] = i2; out[3] = i3;
}

// One thread per keypoint: ids1 -> warp -> ids2 -> 16 penalty columns
__global__ void neigh_kernel(const float* __restrict__ kp1,
                             const float* __restrict__ homo,
                             int* __restrict__ pen) {
  const int idx = blockIdx.x * blockDim.x + threadIdx.x;
  if (idx >= NB * NN) return;
  const int b = idx / NN;
  const float px = kp1[idx * 2 + 0], py = kp1[idx * 2 + 1];
  int ids1[4];
  top4cells(px, py, ids1);
  const float* Hm = homo + b * 9;
  for (int j = 0; j < 4; ++j) {
    const int m = ids1[j];
    const float cx = ((m % NW) + 0.5f) * GS;
    const float cy = ((m / NW) + 0.5f) * GS;
    const float X = Hm[0] * cx + Hm[1] * cy + Hm[2];
    const float Y = Hm[3] * cx + Hm[4] * cy + Hm[5];
    const float Z = Hm[6] * cx + Hm[7] * cy + Hm[8];
    const float inv = 1.0f / (Z + 1e-8f);
    int ids2[4];
    top4cells(X * inv, Y * inv, ids2);
    #pragma unroll
    for (int s = 0; s < 4; ++s) pen[idx * 16 + j * 4 + s] = ids2[s];
  }
}

// Fused WMMA GEMM + penalty + per-row min-4 + hinge loss.
// Block = 128 threads (4 waves) handling a 16-row stripe of one batch.
__global__ void __launch_bounds__(128)
sim_loss_kernel(const unsigned short* __restrict__ kdbf,
                const unsigned short* __restrict__ d2fbf,
                const float* __restrict__ posb,
                const int* __restrict__ pen,
                float* __restrict__ out) {
  __shared__ unsigned short lds_a[16][NC];   // 8 KB : A tile (bf16)
  __shared__ float pos_s[16];
  __shared__ float lds_c[4][16][16];         // 4 KB : per-wave C tile
  __shared__ float mrg[128][4];              // 2 KB : min-4 merge

  const int tid  = threadIdx.x;
  const int wave = tid >> 5;
  const int lane = tid & 31;
  const int b    = blockIdx.y;
  const int row0 = blockIdx.x * 16;

  { // Stage A tile: 16 rows x 256 bf16
    const int r = tid >> 3, seg = tid & 7;   // 32 ushorts per thread
    const unsigned short* src = kdbf + ((size_t)(b * NN + row0 + r)) * NC + seg * 32;
    unsigned short* dst = &lds_a[r][seg * 32];
    #pragma unroll
    for (int q = 0; q < 4; ++q)
      *(u16x8*)(dst + q * 8) = *(const u16x8*)(src + q * 8);
  }
  if (tid < 16) pos_s[tid] = posb[b * NN + row0 + tid];

  const int mrow = lane & 15;                // A row / C col index for this lane
  const int sel  = (lane >> 4) & 1;          // K half selector (ISA A/B layout)
  const int selo = sel * 8;

  // Each sel==0 lane keeps its row's 16 penalty columns in registers.
  int penr[16];
  {
    const int* psrc = pen + ((size_t)(b * NN + row0) + mrow) * 16;
    #pragma unroll
    for (int t = 0; t < 16; ++t) penr[t] = psrc[t];
  }
  __syncthreads();

  float q0 = 3.0e38f, q1 = 3.0e38f, q2 = 3.0e38f, q3 = 3.0e38f;
  const unsigned short* arow = &lds_a[mrow][0];

  for (int tile = wave; tile < NHW / 16; tile += 4) {
    const int colbase = tile * 16;
    const unsigned short* rowp =
        d2fbf + ((size_t)(b * NHW) + colbase + mrow) * NC;
    // prefetch next column tile for this wave (global_prefetch_b8)
    {
      const int nb = min(colbase + 64, NHW - 16);
      __builtin_prefetch((const void*)(d2fbf + ((size_t)(b * NHW) + nb + mrow) * NC), 0, 3);
    }
    v8f acc = {};
    #pragma unroll
    for (int k0 = 0; k0 < NC; k0 += 32) {
      union { v16bf v; u16x8 h[2]; } Af, Bf;
      Af.h[0] = *(const u16x8*)(arow + k0 + selo);
      Af.h[1] = *(const u16x8*)(arow + k0 + 16 + selo);
      Bf.h[0] = *(const u16x8*)(rowp + k0 + selo);
      Bf.h[1] = *(const u16x8*)(rowp + k0 + 16 + selo);
      acc = __builtin_amdgcn_wmma_f32_16x16x32_bf16(
          false, Af.v, false, Bf.v, (short)0, acc, false, false);
    }
    // C layout: element (v, lane) -> row v + 8*sel, col lane&15
    #pragma unroll
    for (int v = 0; v < 8; ++v)
      lds_c[wave][v + sel * 8][mrow] = 2.0f - 2.0f * acc[v];

    // Apply +5 neighborhood penalties into the tile. Single any-hit guard:
    // only ~16 of 300 tiles contain a penalty column at all.
    if (sel == 0) {
      unsigned any = 0;
      #pragma unroll
      for (int t = 0; t < 16; ++t)
        any |= ((unsigned)(penr[t] - colbase) < 16u) ? 1u : 0u;
      if (any) {
        #pragma unroll
        for (int t = 0; t < 16; ++t) {
          const unsigned p = (unsigned)(penr[t] - colbase);
          if (p < 16u) lds_c[wave][mrow][p] += 5.0f;
        }
      }
    }

    // Scan this wave's 16x16 tile: lane handles row=mrow, 8 cols (sel half);
    // branchless min-4 maintenance (v_maxmin_num chains).
    #pragma unroll
    for (int cc = 0; cc < 8; ++cc) {
      const float v = lds_c[wave][mrow][selo + cc];
      min4_insert(v, q0, q1, q2, q3);
    }
  }

  mrg[tid][0] = q0; mrg[tid][1] = q1; mrg[tid][2] = q2; mrg[tid][3] = q3;
  __syncthreads();

  if (tid < 16) {  // merge 8 partial min-4 sets per row, hinge, accumulate
    float r0 = 3.0e38f, r1 = 3.0e38f, r2 = 3.0e38f, r3 = 3.0e38f;
    #pragma unroll
    for (int w = 0; w < 4; ++w) {
      #pragma unroll
      for (int g = 0; g < 2; ++g) {
        const float* s = mrg[w * 32 + g * 16 + tid];
        #pragma unroll
        for (int k = 0; k < 4; ++k) min4_insert(s[k], r0, r1, r2, r3);
      }
    }
    const float p = pos_s[tid] + 1.0f;  // pos + MARGIN
    const float c = fmaxf(p - r0, 0.0f) + fmaxf(p - r1, 0.0f) +
                    fmaxf(p - r2, 0.0f) + fmaxf(p - r3, 0.0f);
    atomicAdd(out, c * (1.0f / (float)(NB * NN * 4)));  // mean * LAMBDA
  }
}

extern "C" void kernel_launch(void* const* d_in, const int* in_sizes, int n_in,
                              void* d_out, int out_size, void* d_ws, size_t ws_size,
                              hipStream_t stream) {
  const float* kp1      = (const float*)d_in[0];
  const float* w_kp1    = (const float*)d_in[1];
  const float* kp1_desc = (const float*)d_in[2];
  const float* desc2    = (const float*)d_in[3];
  const float* homo12   = (const float*)d_in[4];
  float* out = (float*)d_out;

  char* ws = (char*)d_ws;
  unsigned short* kdbf  = (unsigned short*)ws;                                   // 2 MB
  unsigned short* d2fbf = (unsigned short*)(ws + (size_t)NB * NN * NC * 2);      // 9.8 MB
  float* posb = (float*)(ws + (size_t)NB * NN * NC * 2 + (size_t)NB * NHW * NC * 2);
  int*   pen  = (int*)((char*)posb + (size_t)NB * NN * sizeof(float));

  zero_out_kernel<<<1, 32, 0, stream>>>(out);
  prep_kd_pos_kernel<<<(NB * NN) / 8, 256, 0, stream>>>(kp1_desc, w_kp1, desc2, kdbf, posb);
  prep_d2f_kernel<<<(NB * NHW) / 8, 256, 0, stream>>>(desc2, d2fbf);
  neigh_kernel<<<(NB * NN) / 256, 256, 0, stream>>>(kp1, homo12, pen);
  dim3 grid(NN / 16, NB);
  sim_loss_kernel<<<grid, 128, 0, stream>>>(kdbf, d2fbf, posb, pen, out);
}
